// STS_Block_38465727103445
// MI455X (gfx1250) — matrix operand order repaired and checked by reading, not compiled, not run on hardware
//
#include <hip/hip_runtime.h>
#include <hip/hip_bf16.h>

// ---------------------------------------------------------------------------
// STS_Block (Mamba-style SSM) for MI455X / gfx1250, wave32 + WMMA bf16.
//   - All six GEMMs on v_wmma_f32_16x16x32_bf16 with NT-wide register
//     blocking (one A fragment feeds NT WMMAs per K-step).
//   - Selective scan parallelized 96x with a 3-pass chunked linear scan
//     (segment transitions -> sequential combine -> replay with epilogue).
// ---------------------------------------------------------------------------

#define DIMC    96
#define DSTATE  16
#define DINNER  192
#define DTRANK  6
#define BATCH   4
#define LL      9216            // H*W = 96*96
#define BLROWS  (BATCH * LL)    // 36864
#define NPAD    48              // dtbc output cols padded (38 -> 48)
#define NSEG    96              // scan segments per sequence
#define TSEG    (LL / NSEG)     // 96 timesteps per segment
#define NBD     (BATCH * DINNER) // 768 independent scan lanes

typedef __attribute__((ext_vector_type(16))) __bf16          v16bf;
typedef __attribute__((ext_vector_type(16))) unsigned short  v16u;
typedef __attribute__((ext_vector_type(8)))  unsigned short  v8u;
typedef __attribute__((ext_vector_type(8)))  float           v8f;

__device__ __forceinline__ unsigned short f2bf(float f) {
    unsigned int u = __float_as_uint(f);
    unsigned int r = u + 0x7FFFu + ((u >> 16) & 1u);   // round-to-nearest-even
    return (unsigned short)(r >> 16);
}
__device__ __forceinline__ float sigmoidf_(float x) { return 1.f / (1.f + __expf(-x)); }
__device__ __forceinline__ float siluf_(float x)    { return x * sigmoidf_(x); }

// ---- WMMA fragment loaders (ISA 7.12.2, wave32) ---------------------------
__device__ __forceinline__ v16u load_a_u(const unsigned short* base, int lda, int lane) {
    int m = lane & 15, h = lane >> 4;
    const unsigned short* row = base + (size_t)m * lda;
    return *(const v16u*)(row + h * 16);        // K = h*16 .. h*16+15
}
// out = A @ W^T with W (N,K) row-major: column n of B = W[n0+n, k0..k0+31]
__device__ __forceinline__ v16u load_b_u(const unsigned short* W, int ldw,
                                         int n0, int k0, int lane) {
    int n = lane & 15, h = lane >> 4;
    const unsigned short* row = W + (size_t)(n0 + n) * ldw + k0 + h * 16;
    return *(const v16u*)row;                   // 32B aligned, 2x b128
}
__device__ __forceinline__ v8f wmma_bf16(v16u a, v16u b, v8f c) {
    return __builtin_amdgcn_wmma_f32_16x16x32_bf16(
        false, __builtin_bit_cast(v16bf, a),
        false, __builtin_bit_cast(v16bf, b),
        (short)0, c, false, false);
}

// ---- weight conversion ----------------------------------------------------
__global__ void cvt_bf16(const float* __restrict__ src, unsigned short* __restrict__ dst, int n) {
    int i = blockIdx.x * blockDim.x + threadIdx.x;
    if (i < n) dst[i] = f2bf(src[i]);
}
__global__ void cvt_pad_dtbc(const float* __restrict__ src, unsigned short* __restrict__ dst) {
    int i = blockIdx.x * blockDim.x + threadIdx.x;
    if (i >= NPAD * 2 * DINNER) return;
    int r = i / (2 * DINNER), c = i % (2 * DINNER);
    dst[i] = (r < DTRANK + 2 * DSTATE) ? f2bf(src[r * (2 * DINNER) + c]) : (unsigned short)0;
}

// ---- transpose + LayerNorm ------------------------------------------------
__global__ void prep_rows(const float* __restrict__ x, const float* __restrict__ K,
                          const float* __restrict__ Q,
                          const float* __restrict__ kg, const float* __restrict__ kb,
                          const float* __restrict__ qg, const float* __restrict__ qb,
                          unsigned short* __restrict__ xt_bf,
                          unsigned short* __restrict__ kn_bf,
                          unsigned short* __restrict__ qn_bf) {
    int id = blockIdx.x * blockDim.x + threadIdx.x;
    if (id >= BLROWS) return;
    int b = id / LL, l = id % LL;
    size_t pbase = (size_t)b * DIMC * LL + l;
    {
        const float* xb = x + pbase;
        unsigned short* o = xt_bf + (size_t)id * DIMC;
#pragma unroll 4
        for (int c = 0; c < DIMC; c++) o[c] = f2bf(xb[(size_t)c * LL]);
    }
    {
        const float* Kb = K + pbase;
        float s = 0.f, s2 = 0.f;
        for (int c = 0; c < DIMC; c++) { float v = Kb[(size_t)c * LL]; s += v; s2 += v * v; }
        float mean = s * (1.f / DIMC);
        float var  = s2 * (1.f / DIMC) - mean * mean;
        float inv  = rsqrtf(var + 1e-5f);
        unsigned short* o = kn_bf + (size_t)id * DIMC;
        for (int c = 0; c < DIMC; c++)
            o[c] = f2bf((Kb[(size_t)c * LL] - mean) * inv * kg[c] + kb[c]);
    }
    {
        const float* Qb = Q + pbase;
        float s = 0.f, s2 = 0.f;
        for (int c = 0; c < DIMC; c++) { float v = Qb[(size_t)c * LL]; s += v; s2 += v * v; }
        float mean = s * (1.f / DIMC);
        float var  = s2 * (1.f / DIMC) - mean * mean;
        float inv  = rsqrtf(var + 1e-5f);
        unsigned short* o = qn_bf + (size_t)id * DIMC;
        for (int c = 0; c < DIMC; c++)
            o[c] = f2bf((Qb[(size_t)c * LL] - mean) * inv * qg[c] + qb[c]);
    }
}

// ---- NT-blocked WMMA GEMM, f32 out (optional bias) ------------------------
template <int NT>
__global__ void gemm_f32out(const unsigned short* __restrict__ A, int lda,
                            const unsigned short* __restrict__ W, int ldw,
                            const float* __restrict__ bias,
                            float* __restrict__ out, int ldo,
                            int n_groups, int k_tiles, int total_waves) {
    int lane = threadIdx.x & 31;
    int wid  = blockIdx.x * (blockDim.x >> 5) + (threadIdx.x >> 5);
    if (wid >= total_waves) return;
    int tm = wid / n_groups, tg = wid % n_groups;
    int n0 = tg * NT * 16;
    const unsigned short* a_base = A + (size_t)(tm * 16) * lda;
    v8f acc[NT];
#pragma unroll
    for (int j = 0; j < NT; j++) acc[j] = {};
    for (int kt = 0; kt < k_tiles; ++kt) {
        v16u a = load_a_u(a_base + kt * 32, lda, lane);
#pragma unroll
        for (int j = 0; j < NT; j++) {
            v16u b = load_b_u(W, ldw, n0 + j * 16, kt * 32, lane);
            acc[j] = wmma_bf16(a, b, acc[j]);
        }
    }
    int n = lane & 15, h = lane >> 4;
#pragma unroll
    for (int j = 0; j < NT; j++) {
        int col = n0 + j * 16 + n;
        float bv = bias ? bias[col] : 0.f;
#pragma unroll
        for (int r = 0; r < 8; r++) {
            int m = r + 8 * h;
            out[(size_t)(tm * 16 + m) * ldo + col] = acc[j][r] + bv;
        }
    }
}

// ---- NT-blocked WMMA GEMM, bias + SiLU + bf16 out at column offset --------
template <int NT>
__global__ void gemm_silu_bf16out(const unsigned short* __restrict__ A, int lda,
                                  const unsigned short* __restrict__ W, int ldw,
                                  const float* __restrict__ bias,
                                  unsigned short* __restrict__ out, int ldo, int col_off,
                                  int n_groups, int k_tiles, int total_waves) {
    int lane = threadIdx.x & 31;
    int wid  = blockIdx.x * (blockDim.x >> 5) + (threadIdx.x >> 5);
    if (wid >= total_waves) return;
    int tm = wid / n_groups, tg = wid % n_groups;
    int n0 = tg * NT * 16;
    const unsigned short* a_base = A + (size_t)(tm * 16) * lda;
    v8f acc[NT];
#pragma unroll
    for (int j = 0; j < NT; j++) acc[j] = {};
    for (int kt = 0; kt < k_tiles; ++kt) {
        v16u a = load_a_u(a_base + kt * 32, lda, lane);
#pragma unroll
        for (int j = 0; j < NT; j++) {
            v16u b = load_b_u(W, ldw, n0 + j * 16, kt * 32, lane);
            acc[j] = wmma_bf16(a, b, acc[j]);
        }
    }
    int n = lane & 15, h = lane >> 4;
#pragma unroll
    for (int j = 0; j < NT; j++) {
        int col = n0 + j * 16 + n;
        float bv = bias[col];
#pragma unroll
        for (int r = 0; r < 8; r++) {
            int m = r + 8 * h;
            out[(size_t)(tm * 16 + m) * ldo + col_off + col] = f2bf(siluf_(acc[j][r] + bv));
        }
    }
}

// ---- depthwise causal conv (K=4) + SiLU -----------------------------------
__global__ void conv_silu(const float* __restrict__ xin, const float* __restrict__ cw,
                          const float* __restrict__ cb,
                          float* __restrict__ xct,
                          unsigned short* __restrict__ cat1,
                          unsigned short* __restrict__ catz) {
    int idx = blockIdx.x * blockDim.x + threadIdx.x;
    if (idx >= BLROWS * DINNER) return;
    int d = idx % DINNER, r = idx / DINNER;
    int b = r / LL, l = r % LL;
    float acc = cb[d];
#pragma unroll
    for (int k = 0; k < 4; k++) {
        int ll = l - 3 + k;
        if (ll >= 0) acc += xin[((size_t)(b * LL + ll)) * DINNER + d] * cw[d * 4 + k];
    }
    float s = siluf_(acc);
    xct[(size_t)idx] = s;
    unsigned short u = f2bf(s);
    cat1[(size_t)r * (2 * DINNER) + d] = u;
    catz[(size_t)r * (2 * DINNER) + d] = u;
}

// ===========================================================================
// Chunked (3-pass) selective scan.  h_t = a_t*h_{t-1} + x_t  with
// a_t = exp(delta_t*A[n]), x_t = delta_t*u_t*B_t[n].  Per segment:
// pass1: P = prod a_t, S = scan result from h=0.  pass2 (sequential over 96
// segments, 768 threads): h_init per segment.  pass3: replay + y/epilogue.
// ===========================================================================
__device__ __forceinline__ float softplus_(float x) {
    return (x > 20.f) ? x : log1pf(__expf(x));
}

__global__ void scan_pass1(const float* __restrict__ xdbl,  // (BL,48): dt|B|C
                           const float* __restrict__ xct,   // (BL,192) u
                           const float* __restrict__ dt_w, const float* __restrict__ dt_b,
                           const float* __restrict__ A_log,
                           float* __restrict__ Pbuf, float* __restrict__ Sbuf) {
    int gid = blockIdx.x * blockDim.x + threadIdx.x;
    if (gid >= NBD * NSEG) return;
    int id = gid % NBD;                 // consecutive threads -> consecutive d
    int s  = gid / NBD;
    int b = id / DINNER, d = id % DINNER;
    float A[DSTATE], P[DSTATE], c[DSTATE], dtw[DTRANK];
#pragma unroll
    for (int n = 0; n < DSTATE; n++) { A[n] = -__expf(A_log[d * DSTATE + n]); P[n] = 1.f; c[n] = 0.f; }
#pragma unroll
    for (int j = 0; j < DTRANK; j++) dtw[j] = dt_w[d * DTRANK + j];
    float dtb2 = 2.f * dt_b[d];
    int t0 = s * TSEG;
    const float* xr = xdbl + ((size_t)(b * LL + t0)) * NPAD;
    for (int tt = 0; tt < TSEG; tt++) {
        const float* row = xr + (size_t)tt * NPAD;
        __builtin_prefetch(row + NPAD, 0, 0);            // global_prefetch_b8
        float dv = dtb2;
#pragma unroll
        for (int j = 0; j < DTRANK; j++) dv += row[j] * dtw[j];
        float delta = softplus_(dv);
        float u  = xct[((size_t)(b * LL + t0 + tt)) * DINNER + d];
        float du = delta * u;
#pragma unroll
        for (int n = 0; n < DSTATE; n++) {
            float a = __expf(delta * A[n]);
            c[n] = a * c[n] + du * row[DTRANK + n];
            P[n] *= a;
        }
    }
    size_t base = (size_t)gid * DSTATE;
#pragma unroll
    for (int n = 0; n < DSTATE; n++) { Pbuf[base + n] = P[n]; Sbuf[base + n] = c[n]; }
}

// Sequential combine over segments; rewrites Sbuf[s] with h_init(segment s).
__global__ void scan_combine(float* __restrict__ Pbuf, float* __restrict__ Sbuf) {
    int id = blockIdx.x * blockDim.x + threadIdx.x;
    if (id >= NBD) return;
    float h[DSTATE];
#pragma unroll
    for (int n = 0; n < DSTATE; n++) h[n] = 0.f;
    for (int s = 0; s < NSEG; s++) {
        size_t base = ((size_t)s * NBD + id) * DSTATE;
#pragma unroll
        for (int n = 0; n < DSTATE; n++) {
            float Pv = Pbuf[base + n];
            float Sv = Sbuf[base + n];
            Sbuf[base + n] = h[n];          // h_init for this segment
            h[n] = Pv * h[n] + Sv;
        }
    }
}

__global__ void scan_pass3(const float* __restrict__ xdbl,
                           const float* __restrict__ xct,
                           const float* __restrict__ z,
                           const float* __restrict__ dt_w, const float* __restrict__ dt_b,
                           const float* __restrict__ A_log, const float* __restrict__ Dvec,
                           const float* __restrict__ Hinit,   // = Sbuf after combine
                           unsigned short* __restrict__ ybf) {
    int gid = blockIdx.x * blockDim.x + threadIdx.x;
    if (gid >= NBD * NSEG) return;
    int id = gid % NBD;
    int s  = gid / NBD;
    int b = id / DINNER, d = id % DINNER;
    float A[DSTATE], h[DSTATE], dtw[DTRANK];
    size_t hbase = (size_t)gid * DSTATE;
#pragma unroll
    for (int n = 0; n < DSTATE; n++) { A[n] = -__expf(A_log[d * DSTATE + n]); h[n] = Hinit[hbase + n]; }
#pragma unroll
    for (int j = 0; j < DTRANK; j++) dtw[j] = dt_w[d * DTRANK + j];
    float dtb2 = 2.f * dt_b[d];
    float Dd   = Dvec[d];
    int t0 = s * TSEG;
    const float* xr = xdbl + ((size_t)(b * LL + t0)) * NPAD;
    for (int tt = 0; tt < TSEG; tt++) {
        const float* row = xr + (size_t)tt * NPAD;
        __builtin_prefetch(row + NPAD, 0, 0);
        float dv = dtb2;
#pragma unroll
        for (int j = 0; j < DTRANK; j++) dv += row[j] * dtw[j];
        float delta = softplus_(dv);
        size_t ridx = ((size_t)(b * LL + t0 + tt)) * DINNER + d;
        float u  = xct[ridx];
        float du = delta * u;
        float y  = 0.f;
#pragma unroll
        for (int n = 0; n < DSTATE; n++) {
            h[n] = __expf(delta * A[n]) * h[n] + du * row[DTRANK + n];
            y += h[n] * row[DTRANK + DSTATE + n];
        }
        float yv = y + u * Dd;
        float zv = z[ridx];
        ybf[ridx] = f2bf(yv * siluf_(zv));
    }
}

// ---- final GEMM (NT=3) with transposed store to (B,C,H,W) -----------------
__global__ void gemm_out_final(const unsigned short* __restrict__ A,   // (BL,192) bf16
                               const unsigned short* __restrict__ W,   // (96,192) bf16
                               float* __restrict__ out, int total_waves) {
    constexpr int NT = 3;
    int lane = threadIdx.x & 31;
    int wid  = blockIdx.x * (blockDim.x >> 5) + (threadIdx.x >> 5);
    if (wid >= total_waves) return;
    int tm = wid / 2, tg = wid % 2;          // 6 N-tiles = 2 groups of 3
    int n0 = tg * NT * 16;
    const unsigned short* a_base = A + (size_t)(tm * 16) * DINNER;
    v8f acc[NT];
#pragma unroll
    for (int j = 0; j < NT; j++) acc[j] = {};
#pragma unroll
    for (int kt = 0; kt < 6; ++kt) {
        v16u a = load_a_u(a_base + kt * 32, DINNER, lane);
#pragma unroll
        for (int j = 0; j < NT; j++) {
            v16u b = load_b_u(W, DINNER, n0 + j * 16, kt * 32, lane);
            acc[j] = wmma_bf16(a, b, acc[j]);
        }
    }
    int n = lane & 15, h = lane >> 4;
    int gr = tm * 16;                 // 16 | LL so all rows in one batch
    int bb = gr / LL, l0 = gr % LL;
#pragma unroll
    for (int j = 0; j < NT; j++) {
        int c = n0 + j * 16 + n;
        float* col = out + ((size_t)(bb * DIMC + c)) * LL + l0 + 8 * h;
#pragma unroll
        for (int r = 0; r < 8; r++) col[r] = acc[j][r];   // 8 contiguous floats
    }
}

// ---------------------------------------------------------------------------
extern "C" void kernel_launch(void* const* d_in, const int* in_sizes, int n_in,
                              void* d_out, int out_size, void* d_ws, size_t ws_size,
                              hipStream_t stream) {
    const float* x        = (const float*)d_in[0];
    const float* K        = (const float*)d_in[1];
    const float* Q        = (const float*)d_in[2];
    const float* in_proj_w= (const float*)d_in[3];
    const float* conv_w   = (const float*)d_in[4];
    const float* conv_b   = (const float*)d_in[5];
    const float* k_ln_g   = (const float*)d_in[6];
    const float* k_ln_b   = (const float*)d_in[7];
    const float* k_w      = (const float*)d_in[8];
    const float* k_b      = (const float*)d_in[9];
    const float* q_ln_g   = (const float*)d_in[10];
    const float* q_ln_b   = (const float*)d_in[11];
    const float* q_w      = (const float*)d_in[12];
    const float* q_b      = (const float*)d_in[13];
    const float* dtbc_w   = (const float*)d_in[14];
    const float* dt_w     = (const float*)d_in[15];
    const float* dt_b     = (const float*)d_in[16];
    const float* gate_w   = (const float*)d_in[17];
    const float* gate_b   = (const float*)d_in[18];
    const float* A_log    = (const float*)d_in[19];
    const float* Dvec     = (const float*)d_in[20];
    const float* out_w    = (const float*)d_in[21];
    float* out = (float*)d_out;

    // workspace carve-up (256B aligned)
    char* ws = (char*)d_ws;
    size_t off = 0;
    auto carve = [&](size_t bytes) {
        void* p = ws + off;
        off = (off + bytes + 255) & ~(size_t)255;
        return p;
    };
    unsigned short* xt_bf  = (unsigned short*)carve((size_t)BLROWS * DIMC * 2);
    unsigned short* kn_bf  = (unsigned short*)carve((size_t)BLROWS * DIMC * 2);
    unsigned short* qn_bf  = (unsigned short*)carve((size_t)BLROWS * DIMC * 2);
    float*          xin    = (float*)carve((size_t)BLROWS * DINNER * 4);       // reused as z
    unsigned short* cat1   = (unsigned short*)carve((size_t)BLROWS * 2 * DINNER * 2); // [x_t|Kp]
    unsigned short* catz   = (unsigned short*)carve((size_t)BLROWS * 2 * DINNER * 2); // [x_t|Qp]
    float*          xct    = (float*)carve((size_t)BLROWS * DINNER * 4);
    float*          xdbl   = (float*)carve((size_t)BLROWS * NPAD * 4);
    float*          Pbuf   = (float*)carve((size_t)NBD * NSEG * DSTATE * 4);
    float*          Sbuf   = (float*)carve((size_t)NBD * NSEG * DSTATE * 4);
    unsigned short* w_in_bf  = (unsigned short*)carve((size_t)DINNER * DIMC * 2);
    unsigned short* k_w_bf   = (unsigned short*)carve((size_t)DINNER * DIMC * 2);
    unsigned short* q_w_bf   = (unsigned short*)carve((size_t)DINNER * DIMC * 2);
    unsigned short* gate_bf  = (unsigned short*)carve((size_t)DINNER * 2 * DINNER * 2);
    unsigned short* dtbc_bf  = (unsigned short*)carve((size_t)NPAD * 2 * DINNER * 2);
    unsigned short* out_w_bf = (unsigned short*)carve((size_t)DIMC * DINNER * 2);
    float*          zbuf   = xin;                 // alias: xin dead after conv
    unsigned short* ybf    = kn_bf;               // alias: kn|qn contiguous = BL*192 bf16

    // weights -> bf16
    cvt_bf16<<<(DINNER*DIMC + 255)/256, 256, 0, stream>>>(in_proj_w, w_in_bf, DINNER*DIMC);
    cvt_bf16<<<(DINNER*DIMC + 255)/256, 256, 0, stream>>>(k_w,  k_w_bf,  DINNER*DIMC);
    cvt_bf16<<<(DINNER*DIMC + 255)/256, 256, 0, stream>>>(q_w,  q_w_bf,  DINNER*DIMC);
    cvt_bf16<<<(DINNER*2*DINNER + 255)/256, 256, 0, stream>>>(gate_w, gate_bf, DINNER*2*DINNER);
    cvt_bf16<<<(DIMC*DINNER + 255)/256, 256, 0, stream>>>(out_w, out_w_bf, DIMC*DINNER);
    cvt_pad_dtbc<<<(NPAD*2*DINNER + 255)/256, 256, 0, stream>>>(dtbc_w, dtbc_bf);

    // transpose + LN
    prep_rows<<<(BLROWS + 255)/256, 256, 0, stream>>>(x, K, Q, k_ln_g, k_ln_b, q_ln_g, q_ln_b,
                                                      xt_bf, kn_bf, qn_bf);

    const int TM = BLROWS / 16;           // 2304 row tiles
    // xin = xt @ in_proj^T   (M=BL, N=192 -> 3 groups of NT=4, K=96)
    gemm_f32out<4><<<(TM*3)/8, 256, 0, stream>>>(xt_bf, DIMC, w_in_bf, DIMC, nullptr,
                                                 xin, DINNER, 3, 3, TM*3);
    // depthwise conv + SiLU -> xct + bf16 halves of cat1/catz
    conv_silu<<<((size_t)BLROWS*DINNER + 255)/256, 256, 0, stream>>>(xin, conv_w, conv_b,
                                                                     xct, cat1, catz);
    // Kp / Qp  (bias + SiLU, bf16 into cat buffers at col 192)
    gemm_silu_bf16out<4><<<(TM*3)/8, 256, 0, stream>>>(kn_bf, DIMC, k_w_bf, DIMC, k_b,
                                                       cat1, 2*DINNER, DINNER, 3, 3, TM*3);
    gemm_silu_bf16out<4><<<(TM*3)/8, 256, 0, stream>>>(qn_bf, DIMC, q_w_bf, DIMC, q_b,
                                                       catz, 2*DINNER, DINNER, 3, 3, TM*3);
    // x_dbl = [x_t|Kp] @ dtbc^T   (N padded 48 -> 1 group of NT=3, K=384)
    gemm_f32out<3><<<(TM*1)/8, 256, 0, stream>>>(cat1, 2*DINNER, dtbc_bf, 2*DINNER, nullptr,
                                                 xdbl, NPAD, 1, 12, TM*1);
    // z = [x_t|Qp] @ gate^T + gate_b   (K=384) -- writes over dead xin
    gemm_f32out<4><<<(TM*3)/8, 256, 0, stream>>>(catz, 2*DINNER, gate_bf, 2*DINNER, gate_b,
                                                 zbuf, DINNER, 3, 12, TM*3);
    // chunked selective scan (writes ybf over dead kn/qn)
    scan_pass1<<<(NBD*NSEG + 255)/256, 256, 0, stream>>>(xdbl, xct, dt_w, dt_b, A_log,
                                                         Pbuf, Sbuf);
    scan_combine<<<(NBD + 255)/256, 256, 0, stream>>>(Pbuf, Sbuf);
    scan_pass3<<<(NBD*NSEG + 255)/256, 256, 0, stream>>>(xdbl, xct, zbuf, dt_w, dt_b,
                                                         A_log, Dvec, Sbuf, ybf);
    // out = y @ out_w^T, transposed store to (B,C,H,W)
    gemm_out_final<<<(TM*2)/8, 256, 0, stream>>>(ybf, out_w_bf, out, TM*2);
}
